// ContrastiveLoss_83760452206902
// MI455X (gfx1250) — matrix operand order repaired and checked by reading, not compiled, not run on hardware
//
#include <hip/hip_runtime.h>

typedef _Float16 v16h __attribute__((ext_vector_type(16)));
typedef _Float16 v8h  __attribute__((ext_vector_type(8)));
typedef float    v8f  __attribute__((ext_vector_type(8)));
typedef int      v4i  __attribute__((ext_vector_type(4)));

#define NROWS 4096
#define DIM   256
#define TINV  2.0f      // 1 / temperature (0.5)

#define TILE_GBYTES 8192      // 16 rows * 512 B, contiguous in reps
#define ROW_PAD     528       // 512 B row + 16 B pad (bank-conflict-free frags)
#define TILE_LBYTES (16 * ROW_PAD)          // 8448 B per staged tile
#define WAVE_LBYTES (2 * TILE_LBYTES)       // double buffer per wave
#define BLOCK_LBYTES (8 * WAVE_LBYTES)      // 135168 B dynamic LDS per block

// ---------------------------------------------------------------------------
// Kernel 1: L2-normalize rows of emb_i / emb_j, write f16 reps[8192][256].
// ---------------------------------------------------------------------------
__launch_bounds__(256)
__global__ void nt_xent_normalize(const float* __restrict__ emb_i,
                                  const float* __restrict__ emb_j,
                                  _Float16* __restrict__ reps) {
    const int lane = threadIdx.x & 31;
    const int wave = threadIdx.x >> 5;
    const int row  = blockIdx.x * 8 + wave;          // 1024 blocks * 8 waves

    const float* src = (row < NROWS) ? (emb_i + (size_t)row * DIM)
                                     : (emb_j + (size_t)(row - NROWS) * DIM);

    float4 x0 = ((const float4*)src)[lane * 2 + 0];
    float4 x1 = ((const float4*)src)[lane * 2 + 1];

    float ss = x0.x * x0.x + x0.y * x0.y + x0.z * x0.z + x0.w * x0.w
             + x1.x * x1.x + x1.y * x1.y + x1.z * x1.z + x1.w * x1.w;
#pragma unroll
    for (int off = 16; off > 0; off >>= 1)
        ss += __shfl_xor(ss, off, 32);

    const float inv = 1.0f / fmaxf(sqrtf(ss), 1e-12f);

    v8h o;
    o[0] = (_Float16)(x0.x * inv); o[1] = (_Float16)(x0.y * inv);
    o[2] = (_Float16)(x0.z * inv); o[3] = (_Float16)(x0.w * inv);
    o[4] = (_Float16)(x1.x * inv); o[5] = (_Float16)(x1.y * inv);
    o[6] = (_Float16)(x1.z * inv); o[7] = (_Float16)(x1.w * inv);
    *(v8h*)(reps + (size_t)row * DIM + lane * 8) = o;
}

// ---------------------------------------------------------------------------
// Async-copy + wait primitives (CDNA5: GLOBAL_LOAD_ASYNC_TO_LDS_B128,
// S_WAIT_ASYNCCNT).  Builtin path when available, inline-asm fallback.
// ---------------------------------------------------------------------------
__device__ __forceinline__ void async_cp16(void* lds_dst, const void* gsrc) {
#if __has_builtin(__builtin_amdgcn_global_load_async_to_lds_b128)
    typedef __attribute__((address_space(1))) v4i* gp_t;   // global <4 x i32>*
    typedef __attribute__((address_space(3))) v4i* lp_t;   // LDS    <4 x i32>*
    __builtin_amdgcn_global_load_async_to_lds_b128(
        (gp_t)(void*)gsrc, (lp_t)lds_dst, 0, 0);
#else
    unsigned lds_off =
        (unsigned)(unsigned long)(__attribute__((address_space(3))) void*)lds_dst;
    asm volatile("global_load_async_to_lds_b128 %0, %1, off"
                 :: "v"(lds_off), "v"(gsrc) : "memory");
#endif
}

__device__ __forceinline__ void wait_async_le16() {
#if __has_builtin(__builtin_amdgcn_s_wait_asynccnt)
    __builtin_amdgcn_s_wait_asynccnt(16);
#else
    asm volatile("s_wait_asynccnt 0x10" ::: "memory");
#endif
    asm volatile("" ::: "memory");   // keep ds reads behind the wait
}

// ---------------------------------------------------------------------------
// Fused-kernel helpers.
// ---------------------------------------------------------------------------
__device__ __forceinline__ void load_afrag(v16h a[8], const _Float16* reps,
                                           int row0, int lane) {
    // 16-bit A 16x32 layout (contiguous, matches B): lanes 0-15 hold M=lane,
    // halves 0..15 = K k0..k0+15; lanes 16-31 shift by +16.
    const _Float16* arow = reps + (size_t)(row0 + (lane & 15)) * DIM;
    const int hioff = (lane >> 4) * 16;
#pragma unroll
    for (int kf = 0; kf < 8; ++kf) {
        const int k0 = kf * 32 + hioff;
        v8h lo = *(const v8h*)(arow + k0);
        v8h hi = *(const v8h*)(arow + k0 + 8);
        a[kf] = __builtin_shufflevector(lo, hi,
                 0, 1, 2, 3, 4, 5, 6, 7, 8, 9, 10, 11, 12, 13, 14, 15);
    }
}

__device__ __forceinline__ void stage_tile(char* lbuf, const char* gsrc, int lane) {
    // Copy one 8 KB B tile (16 rows x 512 B, contiguous in memory) into LDS
    // with a 16 B per-row pad.  16 async b128 ops, 32 lanes x 16 B each.
    const char* g = gsrc + lane * 16;
    char*       l = lbuf + lane * 16;
#pragma unroll
    for (int r = 0; r < 16; ++r)
        async_cp16(l + r * ROW_PAD, g + r * 512);
}

__device__ __forceinline__ v8f tile_wmma(const v16h a[8], const v16h b[8]) {
    v8f c = {0.f, 0.f, 0.f, 0.f, 0.f, 0.f, 0.f, 0.f};
#pragma unroll
    for (int kf = 0; kf < 8; ++kf)
        c = __builtin_amdgcn_wmma_f32_16x16x32_f16(
                /*neg_a=*/false, a[kf], /*neg_b=*/false, b[kf],
                /*c_mod=*/(short)0, c, /*reuse_a=*/false, /*reuse_b=*/false);
    return c;
}

__device__ __forceinline__ void exp_acc(float accv[8], v8f c) {
#pragma unroll
    for (int r = 0; r < 8; ++r)
        accv[r] += __expf(c[r] * TINV);
}

// ---------------------------------------------------------------------------
// Kernel 2: fused z_i @ reps^T / T -> sum(exp) per row.
// 128 blocks (32 rows each) x 8 waves (64 column tiles each).  Each wave
// keeps TWO 16-row A fragment sets resident (1 load per WMMA of B data) and
// stages its B tiles through a private LDS double buffer with async copies:
// the global stream runs one full tile ahead on ASYNCcnt while WMMAs read
// fragments from LDS at ds latency.  Branch-free hot loop; self/pos logits
// are recomputed as f32 dot products of the f16 rows in the epilogue.
// ---------------------------------------------------------------------------
__launch_bounds__(256)
__global__ void nt_xent_fused(const _Float16* __restrict__ reps,
                              float* __restrict__ per_pair) {
    extern __shared__ char smem[];                 // BLOCK_LBYTES
    __shared__ float s_part[8][32];

    const int lane = threadIdx.x & 31;
    const int wave = threadIdx.x >> 5;
    const int m0   = blockIdx.x * 32;

    v16h a0[8], a1[8];
    load_afrag(a0, reps, m0,      lane);
    load_afrag(a1, reps, m0 + 16, lane);

    float acc0[8], acc1[8];
#pragma unroll
    for (int r = 0; r < 8; ++r) { acc0[r] = 0.0f; acc1[r] = 0.0f; }

    const int jlo = wave * 64;
    const int jhi = jlo + 64;

    char* mybuf = smem + wave * WAVE_LBYTES;
    const char* bimg = (const char*)reps;

    // Prime the pipeline: stage tile jlo into buffer 0.
    stage_tile(mybuf, bimg + (size_t)jlo * TILE_GBYTES, lane);

    // Per-lane fragment base inside a staged tile:
    // lane needs row (lane&15), bytes kf*64 + (lane>>4)*32 .. +31.
    const int frag_off = (lane & 15) * ROW_PAD + (lane >> 4) * 32;

    for (int j = jlo; j < jhi; ++j) {
        // Stage tile j+1 into the other buffer (dummy re-stage of jlo on the
        // last iteration keeps the wait argument uniform).
        const int jn = (j + 1 < jhi) ? (j + 1) : jlo;
        stage_tile(mybuf + ((j + 1) & 1) * TILE_LBYTES,
                   bimg + (size_t)jn * TILE_GBYTES, lane);

        // <=16 outstanding asyncs (in-order completion) => tile j is in LDS.
        wait_async_le16();

        const char* lb = mybuf + (j & 1) * TILE_LBYTES + frag_off;
        v16h b[8];
#pragma unroll
        for (int kf = 0; kf < 8; ++kf)
            b[kf] = *(const v16h*)(lb + kf * 64);

        exp_acc(acc0, tile_wmma(a0, b));
        exp_acc(acc1, tile_wmma(a1, b));
    }

    // Reduce partial sums within each 16-lane half (cols are striped mod 16).
#pragma unroll
    for (int r = 0; r < 8; ++r) {
#pragma unroll
        for (int off = 1; off < 16; off <<= 1) {
            acc0[r] += __shfl_xor(acc0[r], off, 32);
            acc1[r] += __shfl_xor(acc1[r], off, 32);
        }
    }
    if ((lane & 15) == 0) {
        const int half = lane >> 4;      // 0 -> rows 0..7, 1 -> rows 8..15
#pragma unroll
        for (int r = 0; r < 8; ++r) {
            s_part[wave][half * 8 + r]      = acc0[r];
            s_part[wave][16 + half * 8 + r] = acc1[r];
        }
    }
    __syncthreads();

    // Epilogue: threads 0..31 finish one row each.
    if (threadIdx.x < 32) {
        const int m = threadIdx.x;
        float tot = 0.0f;
#pragma unroll
        for (int w = 0; w < 8; ++w) tot += s_part[w][m];

        // self = 2*<z_i,z_i>, pos = 2*<z_i,z_j> directly from the f16 rows.
        const _Float16* zi = reps + (size_t)(m0 + m) * DIM;
        const _Float16* zj = zi + (size_t)NROWS * DIM;
        float sdot = 0.0f, pdot = 0.0f;
#pragma unroll 4
        for (int d = 0; d < DIM; d += 8) {
            v8h xi = *(const v8h*)(zi + d);
            v8h xj = *(const v8h*)(zj + d);
#pragma unroll
            for (int e = 0; e < 8; ++e) {
                const float f = (float)xi[e];
                sdot = fmaf(f, f, sdot);
                pdot = fmaf(f, (float)xj[e], pdot);
            }
        }
        per_pair[m0 + m] = __logf(tot - __expf(sdot * TINV)) - pdot * TINV;
    }
}

// ---------------------------------------------------------------------------
// Kernel 3: deterministic mean of 4096 per-row losses.
// ---------------------------------------------------------------------------
__launch_bounds__(256)
__global__ void nt_xent_reduce(const float* __restrict__ per_pair,
                               float* __restrict__ out) {
    __shared__ float s[256];
    float t = 0.0f;
    for (int i = threadIdx.x; i < NROWS; i += 256) t += per_pair[i];
    s[threadIdx.x] = t;
    __syncthreads();
#pragma unroll
    for (int stride = 128; stride > 0; stride >>= 1) {
        if ((int)threadIdx.x < stride) s[threadIdx.x] += s[threadIdx.x + stride];
        __syncthreads();
    }
    if (threadIdx.x == 0) out[0] = s[0] * (1.0f / NROWS);
}

// ---------------------------------------------------------------------------
extern "C" void kernel_launch(void* const* d_in, const int* in_sizes, int n_in,
                              void* d_out, int out_size, void* d_ws, size_t ws_size,
                              hipStream_t stream) {
    const float* emb_i = (const float*)d_in[0];
    const float* emb_j = (const float*)d_in[1];

    _Float16* reps     = (_Float16*)d_ws;                       // 8192*256*2 = 4 MB
    float*    per_pair = (float*)((char*)d_ws +
                          (size_t)2 * NROWS * DIM * sizeof(_Float16));  // +16 KB

    nt_xent_normalize<<<dim3((2 * NROWS) / 8), dim3(256), 0, stream>>>(emb_i, emb_j, reps);
    nt_xent_fused<<<dim3(NROWS / 32), dim3(256), BLOCK_LBYTES, stream>>>(reps, per_pair);
    nt_xent_reduce<<<dim3(1), dim3(256), 0, stream>>>(per_pair, (float*)d_out);
}
